// Mobile2Former_80753975099796
// MI455X (gfx1250) — matrix operand order repaired and checked by reading, not compiled, not run on hardware
//
#include <hip/hip_runtime.h>

// ---------------------------------------------------------------------------
// CDNA5 (gfx1250) Mobile2Former attention, BF16 WMMA with F32 accumulation.
// ---------------------------------------------------------------------------

typedef __attribute__((ext_vector_type(16))) __bf16 v16bf;  // 32B = 8 VGPRs
typedef __attribute__((ext_vector_type(8)))  __bf16 v8bf;   // 16B
typedef __attribute__((ext_vector_type(8)))  float  v8f;    // 8 VGPRs

#define B_   32
#define C_   512
#define HW_  3136
#define M_   128
#define D_   512
#define INNER_ 2048
#define C2_  256

__device__ __forceinline__ v8f wmma_bf16(v16bf a, v16bf b, v8f c) {
  // D = A(16x32 bf16) x B(32x16 bf16) + C(16x16 f32)
  return __builtin_amdgcn_wmma_f32_16x16x32_bf16(
      /*neg_a=*/false, a, /*neg_b=*/false, b,
      /*c_mod=*/(short)0, c, /*reuse_a=*/false, /*reuse_b=*/false);
}

// A-matrix 16x32 bf16 fragment. rowbase points at (row m, k0).
// lane<16: K = {0..15};  lane>=16: K = {16..31}
__device__ __forceinline__ v16bf load_frag_a(const __bf16* rowbase, int sel) {
  const v8bf lo = *(const v8bf*)(rowbase + 16 * sel);
  const v8bf hi = *(const v8bf*)(rowbase + 16 * sel + 8);
  v16bf f;
#pragma unroll
  for (int i = 0; i < 8; ++i) { f[i] = lo[i]; f[i + 8] = hi[i]; }
  return f;
}

// ---------------------------------------------------------------------------
// x [32][512][3136] f32  ->  xb bf16 (same layout)  +  xt bf16 [32][2][3136][256]
// ---------------------------------------------------------------------------
__global__ __launch_bounds__(256) void conv_x_kernel(
    const float* __restrict__ x, __bf16* __restrict__ xb, __bf16* __restrict__ xt) {
  __shared__ __align__(64) __bf16 tile[64][72];
  const int b  = blockIdx.z;
  const int c0 = blockIdx.y * 64;
  const int n0 = blockIdx.x * 64;
  const int tn = threadIdx.x & 63;
  const int tr = threadIdx.x >> 6;  // 0..3
#pragma unroll
  for (int r = 0; r < 16; ++r) {
    const int cl = r * 4 + tr;
    const size_t idx = ((size_t)b * C_ + c0 + cl) * HW_ + n0 + tn;
    const __bf16 h = (__bf16)x[idx];
    xb[idx] = h;
    tile[cl][tn] = h;
  }
  __syncthreads();
  const int half = c0 >> 8;
  const int c2_0 = c0 & 255;
#pragma unroll
  for (int r = 0; r < 16; ++r) {
    const int nl = r * 4 + tr;
    xt[(((size_t)b * 2 + half) * HW_ + n0 + nl) * C2_ + c2_0 + tn] = tile[tn][nl];
  }
}

__global__ __launch_bounds__(256) void f32_to_bf16_kernel(
    const float* __restrict__ in, __bf16* __restrict__ out, int n) {
  const int i = blockIdx.x * 256 + threadIdx.x;
  if (i < n) out[i] = (__bf16)in[i];
}

// ---------------------------------------------------------------------------
// q[4096][2048] = z[4096][512] @ wq[2048][512]^T + bq      (bf16 out, f32 acc)
// ---------------------------------------------------------------------------
__global__ __launch_bounds__(128) void gemm_q_kernel(
    const __bf16* __restrict__ zb, const __bf16* __restrict__ wqb,
    const float* __restrict__ bq, __bf16* __restrict__ qb) {
  const int wave = threadIdx.x >> 5, lane = threadIdx.x & 31;
  const int nsub = lane & 15, sel = lane >> 4;
  const int rt = blockIdx.x;                 // 0..255
  const int ct = blockIdx.y * 4 + wave;      // 0..127
  const __bf16* arow = zb  + (size_t)(rt * 16 + nsub) * D_;
  const __bf16* brow = wqb + (size_t)(ct * 16 + nsub) * D_ + 16 * sel;
  v8f acc = {};
#pragma unroll
  for (int k = 0; k < 16; ++k)
    acc = wmma_bf16(load_frag_a(arow + k * 32, sel),
                    *(const v16bf*)(brow + k * 32), acc);
  const int col = ct * 16 + nsub;
  const float bias = bq[col];
#pragma unroll
  for (int g = 0; g < 8; ++g)
    qb[(size_t)(rt * 16 + g + 8 * sel) * INNER_ + col] = (__bf16)(acc[g] + bias);
}

// ---------------------------------------------------------------------------
// Flash attention. One block per (b, half, head); 4 waves, 16 query rows/wave.
// K = V = xf[b,half]  (xt: [n][c2] for K-frags, xb: [c][n] for V-frags)
// ---------------------------------------------------------------------------
__global__ __launch_bounds__(128) void attn_kernel(
    const __bf16* __restrict__ qb,   // [32][128][2048]
    const __bf16* __restrict__ xt,   // [32][2][3136][256]
    const __bf16* __restrict__ xb,   // [32][512][3136]
    __bf16* __restrict__ att) {      // [32][128][2048]
  const int b    = blockIdx.y;
  const int half = blockIdx.x >> 3;
  const int head = blockIdx.x & 7;
  const int wave = threadIdx.x >> 5;
  const int lane = threadIdx.x & 31;
  const int nsub = lane & 15;
  const int sel  = lane >> 4;

  __shared__ __align__(64) __bf16 Plds[4][16][32];  // per-wave P staging

  // resident Q fragments (16 rows x 256 features per wave)
  const __bf16* qrow =
      qb + ((size_t)(b * M_ + half * 64 + wave * 16 + nsub)) * INNER_ + head * C2_;
  v16bf qf[8];
#pragma unroll
  for (int k = 0; k < 8; ++k) qf[k] = load_frag_a(qrow + k * 32, sel);

  const __bf16* xt_bh = xt + (((size_t)b * 2 + half) * HW_) * C2_;
  const __bf16* xv    = xb + ((size_t)b * C_ + half * C2_) * (size_t)HW_;

  v8f o[16];
#pragma unroll
  for (int t = 0; t < 16; ++t) o[t] = (v8f){};
  float mrun[8], lrun[8];
#pragma unroll
  for (int g = 0; g < 8; ++g) { mrun[g] = -3.0e38f; lrun[g] = 0.f; }

  const float scale = 0.044194173824159216f;  // 512^-0.5

  for (int n0 = 0; n0 < HW_; n0 += 32) {
    // ---- S = Q @ K^T for a 64x32 block (per wave: two 16x16 tiles) ----
    const __bf16* krow0 = xt_bh + (size_t)(n0 + nsub) * C2_ + 16 * sel;
    const __bf16* krow1 = krow0 + 16 * (size_t)C2_;
    if (n0 + 32 < HW_) {                      // prefetch next key block
      __builtin_prefetch(krow0 + 32 * C2_, 0, 0);
      __builtin_prefetch(krow1 + 32 * C2_, 0, 0);
    }
    v8f s0 = {}, s1 = {};
#pragma unroll
    for (int k = 0; k < 8; ++k) {
      s0 = wmma_bf16(qf[k], *(const v16bf*)(krow0 + k * 32), s0);
      s1 = wmma_bf16(qf[k], *(const v16bf*)(krow1 + k * 32), s1);
    }

    // ---- online softmax (rows live at (g, sel); cols across 16 lanes) ----
#pragma unroll
    for (int g = 0; g < 8; ++g) {
      const float a0 = s0[g] * scale;
      const float a1 = s1[g] * scale;
      float v = fmaxf(a0, a1);
      v = fmaxf(v, __shfl_xor(v, 1));
      v = fmaxf(v, __shfl_xor(v, 2));
      v = fmaxf(v, __shfl_xor(v, 4));
      v = fmaxf(v, __shfl_xor(v, 8));
      const float mnew  = fmaxf(mrun[g], v);
      const float alpha = __expf(mrun[g] - mnew);
      mrun[g] = mnew;
      const float p0 = __expf(a0 - mnew);
      const float p1 = __expf(a1 - mnew);
      float ps = p0 + p1;
      ps += __shfl_xor(ps, 1);
      ps += __shfl_xor(ps, 2);
      ps += __shfl_xor(ps, 4);
      ps += __shfl_xor(ps, 8);
      lrun[g] = lrun[g] * alpha + ps;
      Plds[wave][g + 8 * sel][nsub]      = (__bf16)p0;
      Plds[wave][g + 8 * sel][16 + nsub] = (__bf16)p1;
#pragma unroll
      for (int t = 0; t < 16; ++t) o[t][g] *= alpha;
    }

    // ---- O += P @ V  (P as A-frag from LDS; V B-frags contiguous in xb) ----
    const v16bf pf = load_frag_a(&Plds[wave][nsub][0], sel);
#pragma unroll
    for (int dt = 0; dt < 16; ++dt) {
      const __bf16* vrow = xv + (size_t)(dt * 16 + nsub) * HW_ + n0 + 16 * sel;
      o[dt] = wmma_bf16(pf, *(const v16bf*)vrow, o[dt]);
    }
  }

  // ---- normalize and store: att[b, half*64+row, head*256+d] ----
  __bf16* abase =
      att + ((size_t)(b * M_ + half * 64 + wave * 16)) * INNER_ + head * C2_;
#pragma unroll
  for (int g = 0; g < 8; ++g) {
    const float inv = 1.0f / lrun[g];
    __bf16* ar = abase + (size_t)(g + 8 * sel) * INNER_;
#pragma unroll
    for (int dt = 0; dt < 16; ++dt)
      ar[dt * 16 + nsub] = (__bf16)(o[dt][g] * inv);
  }
}

// ---------------------------------------------------------------------------
// out[4096][512] = z + att[4096][2048] @ wo[512][2048]^T + bo   (f32 out)
// ---------------------------------------------------------------------------
__global__ __launch_bounds__(128) void gemm_out_kernel(
    const __bf16* __restrict__ att, const __bf16* __restrict__ wob,
    const float* __restrict__ bo, const float* __restrict__ z,
    float* __restrict__ out) {
  const int wave = threadIdx.x >> 5, lane = threadIdx.x & 31;
  const int nsub = lane & 15, sel = lane >> 4;
  const int rt = blockIdx.x;                 // 0..255
  const int ct = blockIdx.y * 4 + wave;      // 0..31
  const __bf16* arow = att + (size_t)(rt * 16 + nsub) * INNER_;
  const __bf16* brow = wob + (size_t)(ct * 16 + nsub) * INNER_ + 16 * sel;
  v8f acc = {};
#pragma unroll
  for (int k = 0; k < 64; ++k)
    acc = wmma_bf16(load_frag_a(arow + k * 32, sel),
                    *(const v16bf*)(brow + k * 32), acc);
  const int col = ct * 16 + nsub;
  const float bias = bo[col];
#pragma unroll
  for (int g = 0; g < 8; ++g) {
    const size_t row = (size_t)(rt * 16 + g + 8 * sel);
    out[row * D_ + col] = acc[g] + bias + z[row * D_ + col];
  }
}

// ---------------------------------------------------------------------------
extern "C" void kernel_launch(void* const* d_in, const int* in_sizes, int n_in,
                              void* d_out, int out_size, void* d_ws, size_t ws_size,
                              hipStream_t stream) {
  (void)in_sizes; (void)n_in; (void)out_size; (void)ws_size;
  const float* x  = (const float*)d_in[0];
  const float* z  = (const float*)d_in[1];
  const float* wq = (const float*)d_in[2];
  const float* bq = (const float*)d_in[3];
  const float* wo = (const float*)d_in[4];
  const float* bo = (const float*)d_in[5];
  float* out = (float*)d_out;

  // workspace carve-up (bf16 elements)
  const size_t N_XB  = (size_t)B_ * C_ * HW_;      // 51,380,224
  const size_t N_XT  = N_XB;
  const size_t N_Q   = (size_t)B_ * M_ * INNER_;   // 8,388,608
  const size_t N_Z   = (size_t)B_ * M_ * D_;       // 2,097,152
  const size_t N_WQ  = (size_t)INNER_ * D_;        // 1,048,576
  const size_t N_WO  = N_WQ;
  const size_t N_ATT = N_Q;

  __bf16* ws  = (__bf16*)d_ws;
  __bf16* xb  = ws;             ws += N_XB;
  __bf16* xt  = ws;             ws += N_XT;
  __bf16* qb  = ws;             ws += N_Q;
  __bf16* zb  = ws;             ws += N_Z;
  __bf16* wqb = ws;             ws += N_WQ;
  __bf16* wob = ws;             ws += N_WO;
  __bf16* attb = ws;            ws += N_ATT;

  // 1) precision conversion / layout
  conv_x_kernel<<<dim3(HW_ / 64, C_ / 64, B_), 256, 0, stream>>>(x, xb, xt);
  f32_to_bf16_kernel<<<(int)((N_Z  + 255) / 256), 256, 0, stream>>>(z,  zb,  (int)N_Z);
  f32_to_bf16_kernel<<<(int)((N_WQ + 255) / 256), 256, 0, stream>>>(wq, wqb, (int)N_WQ);
  f32_to_bf16_kernel<<<(int)((N_WO + 255) / 256), 256, 0, stream>>>(wo, wob, (int)N_WO);

  // 2) q = z @ wq^T + bq
  gemm_q_kernel<<<dim3(B_ * M_ / 16, INNER_ / 16 / 4), 128, 0, stream>>>(zb, wqb, bq, qb);

  // 3) flash attention per (b, half, head)
  attn_kernel<<<dim3(16, B_), 128, 0, stream>>>(qb, xt, xb, attb);

  // 4) out = z + att @ wo^T + bo
  gemm_out_kernel<<<dim3(B_ * M_ / 16, D_ / 16 / 4), 128, 0, stream>>>(attb, wob, bo, z, out);
}
